// MambaBlock_489626272166
// MI455X (gfx1250) — compile-verified, matrix operand order checked by reading, not run on hardware
//
#include <hip/hip_runtime.h>
#include <hip/hip_bf16.h>
#include <math.h>
#include <stdint.h>

typedef __attribute__((ext_vector_type(16))) _Float16 v16h;
typedef __attribute__((ext_vector_type(8)))  _Float16 v8h;
typedef __attribute__((ext_vector_type(8)))  float    v8f;
typedef __attribute__((ext_vector_type(4)))  unsigned int u32x4;

#define DIMM   1024
#define DIN    2048
#define R2     4096   // 2*DIN
#define NST    16
#define DTR    64
#define GPROJ  96     // DTR + 2*NST
#define BB     2
#define LL     1024
#define ME     (BB*LL)   // 2048 rows (b*l)

// ---------------- WMMA GEMM: C[M,N] = A[M,K] * W[N,K]^T (f16 in, f32 out) ---
#define BM 128
#define BN 128
#define BK 32
#define LDP (BK + 8)   // padded LDS row (halves): 80B stride, 16B aligned

__device__ __forceinline__ void wait_async0() {
#if __has_builtin(__builtin_amdgcn_s_wait_asynccnt)
    __builtin_amdgcn_s_wait_asynccnt(0);
#else
    asm volatile("s_wait_asynccnt 0" ::: "memory");
#endif
}

// per-lane 16B async copy global -> LDS (gfx1250 GLOBAL_LOAD_ASYNC_TO_LDS_B128)
__device__ __forceinline__ void async_copy_b128(const void* gptr, void* lptr) {
    uint32_t lds  = (uint32_t)(uintptr_t)lptr;     // low 32 bits of flat LDS ptr = LDS offset
    uint64_t addr = (uint64_t)(uintptr_t)gptr;
    asm volatile("global_load_async_to_lds_b128 %0, %1, off"
                 :: "v"(lds), "v"(addr) : "memory");
}

__global__ __launch_bounds__(256) void gemm_awt_f16(
    const _Float16* __restrict__ A, const _Float16* __restrict__ W,
    float* __restrict__ C, int M, int N, int K)
{
    __shared__ __align__(16) _Float16 sA[2][BM][LDP];
    __shared__ __align__(16) _Float16 sB[2][BN][LDP];

    const int tid   = threadIdx.x;
    const int lane  = tid & 31;
    const int wave  = tid >> 5;
    const int lhalf = lane & 15;         // row (A) / col (B/C) within 16
    const int lhi   = lane >> 4;         // half-wave select
    const int wm    = (wave >> 1) * 32;  // 4 waves along M: 0,32,64,96
    const int wn    = (wave & 1) * 64;   // 2 waves along N: 0,64
    const int bm    = blockIdx.x * BM;
    const int bn    = blockIdx.y * BN;

    // staging geometry: each tile is 512 x 16B chunks; thread owns chunks tid, tid+256
    const int r0 = tid >> 2;             // rows 0..63
    const int c0 = tid & 3;              // 16B chunk within 32-half row

    // zero-fill out-of-range B rows once: never overwritten by async loads,
    // so they stay zero for every K chunk (N=96 edge case)
    #pragma unroll
    for (int t = 0; t < 2; ++t) {
        int r = r0 + t * 64;
        if (bn + r >= N) {
            *(u32x4*)&sB[0][r][c0 * 8] = (u32x4){0u, 0u, 0u, 0u};
            *(u32x4*)&sB[1][r][c0 * 8] = (u32x4){0u, 0u, 0u, 0u};
        }
    }

    v8f acc[2][4] = {};                  // 2 (M) x 4 (N) tiles of 16x16 f32

    // ---- async stage of one K-chunk into LDS buffer `buf` ----
    auto stage = [&](int buf, int k0) {
        #pragma unroll
        for (int t = 0; t < 2; ++t) {
            int r = r0 + t * 64;
            async_copy_b128(A + (size_t)(bm + r) * K + k0 + c0 * 8,
                            &sA[buf][r][c0 * 8]);
        }
        #pragma unroll
        for (int t = 0; t < 2; ++t) {
            int r = r0 + t * 64;
            if (bn + r < N)
                async_copy_b128(W + (size_t)(bn + r) * K + k0 + c0 * 8,
                                &sB[buf][r][c0 * 8]);
        }
    };

    stage(0, 0);
    int buf = 0;
    for (int k0 = 0; k0 < K; k0 += BK) {
        wait_async0();            // my chunks of `buf` landed in LDS
        __syncthreads();          // everyone's chunks landed; prev reads of buf^1 done
        if (k0 + BK < K) stage(buf ^ 1, k0 + BK);   // overlap next fetch with math

        // fragments: lane<16 holds K {0..7,16..23}; lane>=16 holds K {8..15,24..31}
        v16h af[2], bf[4];
        #pragma unroll
        for (int i = 0; i < 2; ++i) {
            const _Float16* p = &sA[buf][wm + i * 16 + lhalf][lhi * 8];
            v8h lo = *(const v8h*)p;
            v8h hi = *(const v8h*)(p + 16);
            af[i] = __builtin_shufflevector(lo, hi, 0,1,2,3,4,5,6,7,8,9,10,11,12,13,14,15);
        }
        #pragma unroll
        for (int j = 0; j < 4; ++j) {
            const _Float16* p = &sB[buf][wn + j * 16 + lhalf][lhi * 8];
            v8h lo = *(const v8h*)p;
            v8h hi = *(const v8h*)(p + 16);
            bf[j] = __builtin_shufflevector(lo, hi, 0,1,2,3,4,5,6,7,8,9,10,11,12,13,14,15);
        }
        #pragma unroll
        for (int i = 0; i < 2; ++i)
            #pragma unroll
            for (int j = 0; j < 4; ++j)
                acc[i][j] = __builtin_amdgcn_wmma_f32_16x16x32_f16(
                    false, af[i], false, bf[j], (short)0, acc[i][j], false, false);
        buf ^= 1;
    }

    // store: VGPR r holds M=r (lanes 0-15) / M=r+8 (lanes 16-31), N=lane&15
    #pragma unroll
    for (int i = 0; i < 2; ++i)
        #pragma unroll
        for (int j = 0; j < 4; ++j) {
            int col = bn + wn + j * 16 + lhalf;
            if (col < N) {
                #pragma unroll
                for (int r = 0; r < 8; ++r) {
                    int row = bm + wm + i * 16 + lhi * 8 + r;
                    C[(size_t)row * N + col] = acc[i][j][r];
                }
            }
        }
}

// ---------------- elementwise kernels ----------------
__global__ void f32_to_f16_kernel(const float* __restrict__ s,
                                  _Float16* __restrict__ d, int n)
{
    int i = blockIdx.x * blockDim.x + threadIdx.x;
    if (i < n) d[i] = (_Float16)s[i];
}

// causal depthwise conv1d (K=4) + bias + SiLU over xr[:, 0:DIN]
__global__ void conv_silu_kernel(const float* __restrict__ xr,
                                 const float* __restrict__ cw,
                                 const float* __restrict__ cb,
                                 float* __restrict__ u32,
                                 _Float16* __restrict__ u16)
{
    int idx = blockIdx.x * blockDim.x + threadIdx.x;   // over ME*DIN
    if (idx >= ME * DIN) return;
    int d   = idx & (DIN - 1);
    int row = idx >> 11;          // DIN = 2^11
    int l   = row & (LL - 1);
    float acc = cb[d];
    #pragma unroll
    for (int t = 0; t < 4; ++t) {
        if (l - t >= 0)
            acc += cw[d * 4 + (3 - t)] * xr[(size_t)(row - t) * R2 + d];
    }
    float s = acc / (1.0f + __expf(-acc));   // SiLU
    u32[(size_t)row * DIN + d] = s;
    u16[(size_t)row * DIN + d] = (_Float16)s;
}

// slice first DTR cols of x_dbl -> f16 (input to dt_proj GEMM)
__global__ void slice_dlt_kernel(const float* __restrict__ xdbl,
                                 _Float16* __restrict__ dlt16)
{
    int idx = blockIdx.x * blockDim.x + threadIdx.x;   // over ME*DTR
    if (idx >= ME * DTR) return;
    int row = idx / DTR;
    int c   = idx - row * DTR;
    dlt16[idx] = (_Float16)xdbl[(size_t)row * GPROJ + c];
}

// selective scan: one thread per (b,d); fuses softplus(delta), state update,
// C-contraction, D skip, and res SiLU gate; writes gated y as f16
__global__ __launch_bounds__(256) void scan_kernel(
    const float* __restrict__ dRaw,   // [ME, DIN] pre-bias delta
    const float* __restrict__ u32,    // [ME, DIN]
    const float* __restrict__ xdbl,   // [ME, GPROJ] (B at 64..79, C at 80..95)
    const float* __restrict__ xr,     // [ME, R2]  (res = cols DIN..)
    const float* __restrict__ A_log,  // [DIN, NST]
    const float* __restrict__ b_dt,   // [DIN]
    const float* __restrict__ Dp,     // [DIN]
    _Float16* __restrict__ yg16)      // [ME, DIN]
{
    int gid = blockIdx.x * blockDim.x + threadIdx.x;   // 0..BB*DIN-1
    if (gid >= BB * DIN) return;
    int b = gid >> 11;
    int d = gid & (DIN - 1);

    float Ac[NST];
    #pragma unroll
    for (int n = 0; n < NST; ++n) Ac[n] = -__expf(A_log[d * NST + n]);
    float h[NST];
    #pragma unroll
    for (int n = 0; n < NST; ++n) h[n] = 0.0f;

    const float bd = b_dt[d];
    const float Dd = Dp[d];

    for (int l = 0; l < LL; ++l) {
        int row = b * LL + l;
        float dr = dRaw[(size_t)row * DIN + d] + bd;
        float delta = (dr > 20.0f) ? dr : log1pf(__expf(dr));   // softplus
        float uu = u32[(size_t)row * DIN + d];
        float du = delta * uu;
        const float* bc = xdbl + (size_t)row * GPROJ;
        float y = uu * Dd;
        #pragma unroll
        for (int n = 0; n < NST; ++n) {
            float dA = __expf(delta * Ac[n]);
            h[n] = dA * h[n] + du * bc[DTR + n];       // h = dA*h + dBu
            y   += h[n] * bc[DTR + NST + n];           // y += h*C
        }
        float r  = xr[(size_t)row * R2 + DIN + d];     // res
        float yg = y * (r / (1.0f + __expf(-r)));      // y * silu(res)
        yg16[(size_t)row * DIN + d] = (_Float16)yg;
    }
}

// ---------------- host side ----------------
extern "C" void kernel_launch(void* const* d_in, const int* in_sizes, int n_in,
                              void* d_out, int out_size, void* d_ws, size_t ws_size,
                              hipStream_t stream)
{
    const float* x      = (const float*)d_in[0];
    const float* W_in   = (const float*)d_in[1];
    const float* conv_w = (const float*)d_in[2];
    const float* conv_b = (const float*)d_in[3];
    const float* W_x    = (const float*)d_in[4];
    const float* W_dt   = (const float*)d_in[5];
    const float* b_dt   = (const float*)d_in[6];
    const float* A_log  = (const float*)d_in[7];
    const float* Dp     = (const float*)d_in[8];
    const float* W_out  = (const float*)d_in[9];
    float* out = (float*)d_out;

    char* w = (char*)d_ws;
    size_t off = 0;
    auto alloc = [&](size_t bytes) -> void* {
        void* p = w + off;
        off += (bytes + 255) & ~(size_t)255;
        return p;
    };

    _Float16* x16    = (_Float16*)alloc((size_t)ME * DIMM * 2);
    _Float16* Win16  = (_Float16*)alloc((size_t)R2 * DIMM * 2);
    _Float16* Wx16   = (_Float16*)alloc((size_t)GPROJ * DIN * 2);
    _Float16* Wdt16  = (_Float16*)alloc((size_t)DIN * DTR * 2);
    _Float16* Wout16 = (_Float16*)alloc((size_t)DIMM * DIN * 2);
    float*    xr     = (float*)   alloc((size_t)ME * R2 * 4);
    float*    u32    = (float*)   alloc((size_t)ME * DIN * 4);
    _Float16* u16    = (_Float16*)alloc((size_t)ME * DIN * 2);
    float*    xdbl   = (float*)   alloc((size_t)ME * GPROJ * 4);
    _Float16* dlt16  = (_Float16*)alloc((size_t)ME * DTR * 2);
    float*    dRaw   = (float*)   alloc((size_t)ME * DIN * 4);
    _Float16* yg16   = (_Float16*)alloc((size_t)ME * DIN * 2);
    (void)ws_size; (void)in_sizes; (void)n_in; (void)out_size;

    const int T = 256;
    auto blocks = [](int n, int t) { return (n + t - 1) / t; };

    // f32 -> f16 conversions for GEMM operands
    f32_to_f16_kernel<<<blocks(ME * DIMM, T), T, 0, stream>>>(x,     x16,    ME * DIMM);
    f32_to_f16_kernel<<<blocks(R2 * DIMM, T), T, 0, stream>>>(W_in,  Win16,  R2 * DIMM);
    f32_to_f16_kernel<<<blocks(GPROJ * DIN, T), T, 0, stream>>>(W_x,  Wx16,  GPROJ * DIN);
    f32_to_f16_kernel<<<blocks(DIN * DTR, T), T, 0, stream>>>(W_dt,  Wdt16,  DIN * DTR);
    f32_to_f16_kernel<<<blocks(DIMM * DIN, T), T, 0, stream>>>(W_out, Wout16, DIMM * DIN);

    // in_proj: xr[ME,R2] = x16[ME,DIMM] @ Win16[R2,DIMM]^T
    gemm_awt_f16<<<dim3(ME / BM, (R2 + BN - 1) / BN), T, 0, stream>>>(
        x16, Win16, xr, ME, R2, DIMM);

    // depthwise causal conv + SiLU on u half of xr
    conv_silu_kernel<<<blocks(ME * DIN, T), T, 0, stream>>>(xr, conv_w, conv_b, u32, u16);

    // x_proj: xdbl[ME,96] = u16 @ Wx16^T  (N=96 masked inside tile)
    gemm_awt_f16<<<dim3(ME / BM, (GPROJ + BN - 1) / BN), T, 0, stream>>>(
        u16, Wx16, xdbl, ME, GPROJ, DIN);

    // slice dt_rank cols -> f16
    slice_dlt_kernel<<<blocks(ME * DTR, T), T, 0, stream>>>(xdbl, dlt16);

    // dt_proj: dRaw[ME,DIN] = dlt16[ME,DTR] @ Wdt16[DIN,DTR]^T
    gemm_awt_f16<<<dim3(ME / BM, (DIN + BN - 1) / BN), T, 0, stream>>>(
        dlt16, Wdt16, dRaw, ME, DIN, DTR);

    // selective scan + gating -> yg16
    scan_kernel<<<blocks(BB * DIN, T), T, 0, stream>>>(
        dRaw, u32, xdbl, xr, A_log, b_dt, Dp, yg16);

    // out_proj: out[ME,DIMM] = yg16 @ Wout16^T
    gemm_awt_f16<<<dim3(ME / BM, (DIMM + BN - 1) / BN), T, 0, stream>>>(
        yg16, Wout16, out, ME, DIMM, DIN);
}
